// Block_42167988912800
// MI455X (gfx1250) — matrix operand order repaired and checked by reading, not compiled, not run on hardware
//
#include <hip/hip_runtime.h>
#include <math.h>

// ---------------------------------------------------------------------------
// CDNA5 / gfx1250, wave32.  Matmuls: v_wmma_f32_16x16x32_bf16.
// GEMM global->LDS staging: Tensor Data Mover (tensor_load_to_lds) with
// double-buffered LDS, one barrier per K-step.
// ---------------------------------------------------------------------------

typedef __bf16 bf16_t;
typedef __attribute__((ext_vector_type(8)))  __bf16 bf16x8;
typedef __attribute__((ext_vector_type(16))) __bf16 bf16x16;
typedef __attribute__((ext_vector_type(8)))  float  f32x8;
typedef __attribute__((ext_vector_type(4)))  unsigned int u32x4;
typedef __attribute__((ext_vector_type(4)))  int          i32x4;
typedef __attribute__((ext_vector_type(8)))  int          i32x8;

#define BB     16
#define SEQ    1024
#define DMODEL 512
#define NHEAD  4
#define HDIM   128
#define MLPD   1024
#define NTOK   (BB * SEQ)          // 16384 tokens

#if __has_builtin(__builtin_amdgcn_tensor_load_to_lds) && \
    __has_builtin(__builtin_amdgcn_s_wait_tensorcnt)
#define USE_TDM 1
#else
#define USE_TDM 0
#endif

// --- WMMA fragment helpers --------------------------------------------------
// A fragment (16x32 bf16, M x K):  lane m = lane&15, kgroup = lane>>4
//   elems 0..7  = K[kgroup*8 .. +7],  elems 8..15 = K[16+kgroup*8 .. +7]
__device__ __forceinline__ bf16x16 ld_frag_a(const bf16_t* rowp, int hi) {
  bf16x8 l0 = *(const bf16x8*)(rowp + hi * 8);
  bf16x8 l1 = *(const bf16x8*)(rowp + 16 + hi * 8);
  return __builtin_shufflevector(l0, l1, 0,1,2,3,4,5,6,7,8,9,10,11,12,13,14,15);
}
// B fragment (32x16 bf16, K x N): lane n = lane&15, half = lane>>4
//   elems 0..7  = K[half*8 .. +7],  elems 8..15 = K[16+half*8 .. +7]
__device__ __forceinline__ bf16x16 ld_frag_b(const bf16_t* colp, int hi) {
  bf16x8 l0 = *(const bf16x8*)(colp + hi * 8);
  bf16x8 l1 = *(const bf16x8*)(colp + 16 + hi * 8);
  return __builtin_shufflevector(l0, l1, 0,1,2,3,4,5,6,7,8,9,10,11,12,13,14,15);
}
__device__ __forceinline__ f32x8 wmma_bf16(bf16x16 a, bf16x16 b, f32x8 c) {
  return __builtin_amdgcn_wmma_f32_16x16x32_bf16(false, a, false, b, (short)0, c,
                                                 false, false);
}

#if USE_TDM
// --- TDM: DMA one (rows x 64B) tile, row stride = strideElems bf16, into LDS
// with 16B of padding appended per 64B row  ->  LDS row stride = 40 bf16.
__device__ __forceinline__ void tdm_load_tile(const bf16_t* gsrc, int strideElems,
                                              bf16_t* ldst, int rows) {
  unsigned lds = (unsigned)(unsigned long long)(size_t)ldst;     // low 32 = LDS offset
  unsigned long long ga = (unsigned long long)(size_t)gsrc;
  u32x4 g0 = { 1u,                                   // count=1 (valid descriptor)
               lds,                                  // lds_addr
               (unsigned)ga,                         // global_addr[31:0]
               (unsigned)(ga >> 32) | (2u << 30) };  // global_addr[56:32] | type=2
  const unsigned td0 = 1u << 20, td1 = 1u << 20;     // huge tensor dims: no clip
  unsigned long long s0 = (unsigned long long)strideElems * 2ull / 8ull; // 8B units
  i32x8 g1;
  // dw0: wg_mask=0 | data_size=3(8B)<<16 | pad_en<<20 | pad_interval=3(16DW)<<22
  //      | pad_amount=3(4DW=16B)<<25
  g1[0] = (3 << 16) | (1 << 20) | (3 << 22) | (3 << 25);
  g1[1] = (int)((td0 & 0xffffu) << 16);                        // barrier_addr=0 | td0.lo
  g1[2] = (int)((td0 >> 16) | ((td1 & 0xffffu) << 16));        // td0.hi | td1.lo
  g1[3] = (int)((td1 >> 16) | (8u << 16));                     // td1.hi | tile_dim0=8
  g1[4] = (int)(rows & 0xffff);                                // tile_dim1 | tile_dim2=0
  g1[5] = (int)(unsigned)(s0 & 0xffffffffu);                   // dim0_stride[31:0]
  g1[6] = (int)(unsigned)((s0 >> 32) & 0xffffu);               // stride hi | dim1_stride=0
  g1[7] = 0;
  i32x4 z4 = { 0, 0, 0, 0 };
#if __clang_major__ >= 23
  i32x8 z8 = { 0, 0, 0, 0, 0, 0, 0, 0 };
  __builtin_amdgcn_tensor_load_to_lds(g0, g1, z4, z4, z8, 0);
#else
  __builtin_amdgcn_tensor_load_to_lds(g0, g1, z4, z4, 0);
#endif
}
#endif

// --- weight convert: W[K,N] f32 (row major)  ->  Wt[N,K] bf16 ---------------
__global__ void wcvt_kernel(const float* __restrict__ W, bf16_t* __restrict__ Wt,
                            int K, int N) {
  int idx = blockIdx.x * 256 + threadIdx.x;
  if (idx < K * N) {
    int k = idx / N, n = idx - k * N;
    Wt[(size_t)n * K + k] = (bf16_t)W[idx];
  }
}

// --- LayerNorm over D=512, one block (256 thr) per token --------------------
__global__ __launch_bounds__(256) void ln_kernel(const float* __restrict__ x,
                                                 const float* __restrict__ g,
                                                 const float* __restrict__ bta,
                                                 float* __restrict__ outf,
                                                 bf16_t* __restrict__ outb) {
  int row = blockIdx.x;
  const float* xr = x + (size_t)row * DMODEL;
  int tid = threadIdx.x;
  float a0 = xr[tid], a1 = xr[tid + 256];
  float s = a0 + a1, ss = a0 * a0 + a1 * a1;
#pragma unroll
  for (int off = 16; off; off >>= 1) {
    s += __shfl_xor(s, off, 32);
    ss += __shfl_xor(ss, off, 32);
  }
  __shared__ float red[16];
  if ((tid & 31) == 0) { red[tid >> 5] = s; red[8 + (tid >> 5)] = ss; }
  __syncthreads();
  float st = 0.f, sst = 0.f;
#pragma unroll
  for (int i = 0; i < 8; ++i) { st += red[i]; sst += red[8 + i]; }
  float mean = st * (1.f / DMODEL);
  float var  = sst * (1.f / DMODEL) - mean * mean;
  float inv  = rsqrtf(var + 1e-6f);
  float y0 = (a0 - mean) * inv * g[tid]       + bta[tid];
  float y1 = (a1 - mean) * inv * g[tid + 256] + bta[tid + 256];
  size_t o = (size_t)row * DMODEL + tid;
  if (outf) { outf[o] = y0; outf[o + 256] = y1; }
  if (outb) { outb[o] = (bf16_t)y0; outb[o + 256] = (bf16_t)y1; }
}

// --- GEMM:  C[M,N] = A[M,K](bf16) * Bt[N,K](bf16)^T + bias (+res)(+gelu) ----
// block 256 thr (8 waves), tile 128x128, waves 4(M) x 2(N), wave tile 32x64.
// Double-buffered LDS; TDM DMA for tile staging when available.
__global__ __launch_bounds__(256) void gemm_kernel(
    const bf16_t* __restrict__ A, const bf16_t* __restrict__ Bt,
    const float* __restrict__ bias, const float* __restrict__ residual,
    float* __restrict__ outf, bf16_t* __restrict__ outb,
    int M, int N, int K, int act_gelu) {
  __shared__ alignas(16) bf16_t As[2][128 * 40];
  __shared__ alignas(16) bf16_t Bs[2][128 * 40];
  int tid = threadIdx.x, lane = tid & 31, wave = tid >> 5;
  int hi = lane >> 4, lo = lane & 15;
  int wm = wave & 3, wn = wave >> 2;
  int m0 = blockIdx.x * 128, n0 = blockIdx.y * 128;

  f32x8 acc[2][4];
#pragma unroll
  for (int mt = 0; mt < 2; ++mt)
#pragma unroll
    for (int nt = 0; nt < 4; ++nt)
#pragma unroll
      for (int r = 0; r < 8; ++r) acc[mt][nt][r] = 0.f;

#if USE_TDM
  // prime buffer 0 (TDM issued by wave 0 only; TENSORcnt is per-wave)
  if (wave == 0) {
    tdm_load_tile(A  + (size_t)m0 * K, K, &As[0][0], 128);
    tdm_load_tile(Bt + (size_t)n0 * K, K, &Bs[0][0], 128);
    __builtin_amdgcn_s_wait_tensorcnt(0);
  }
  __syncthreads();
  int buf = 0;
  for (int k0 = 0; k0 < K; k0 += 32) {
    if (wave == 0 && k0 + 32 < K) {   // DMA next K-tile into the other buffer
      tdm_load_tile(A  + (size_t)m0 * K + k0 + 32, K, &As[buf ^ 1][0], 128);
      tdm_load_tile(Bt + (size_t)n0 * K + k0 + 32, K, &Bs[buf ^ 1][0], 128);
    }
#pragma unroll
    for (int mt = 0; mt < 2; ++mt) {
      bf16x16 a = ld_frag_a(&As[buf][(wm * 32 + mt * 16 + lo) * 40], hi);
#pragma unroll
      for (int nt = 0; nt < 4; ++nt) {
        bf16x16 b = ld_frag_b(&Bs[buf][(wn * 64 + nt * 16 + lo) * 40], hi);
        acc[mt][nt] = wmma_bf16(a, b, acc[mt][nt]);
      }
    }
    if (wave == 0) __builtin_amdgcn_s_wait_tensorcnt(0);
    __syncthreads();
    buf ^= 1;
  }
#else
  // fallback: synchronous staged copies, double-buffered, 1 barrier / K-step
  {
    int c0 = tid, c1 = 256 + tid;
    bf16x8 a0 = *(const bf16x8*)&A [(size_t)(m0 + (c0 >> 2)) * K + (c0 & 3) * 8];
    bf16x8 b0 = *(const bf16x8*)&Bt[(size_t)(n0 + (c0 >> 2)) * K + (c0 & 3) * 8];
    bf16x8 a1 = *(const bf16x8*)&A [(size_t)(m0 + (c1 >> 2)) * K + (c1 & 3) * 8];
    bf16x8 b1 = *(const bf16x8*)&Bt[(size_t)(n0 + (c1 >> 2)) * K + (c1 & 3) * 8];
    int buf = 0;
    for (int k0 = 0; k0 < K; k0 += 32) {
      *(bf16x8*)&As[buf][(c0 >> 2) * 40 + (c0 & 3) * 8] = a0;
      *(bf16x8*)&Bs[buf][(c0 >> 2) * 40 + (c0 & 3) * 8] = b0;
      *(bf16x8*)&As[buf][(c1 >> 2) * 40 + (c1 & 3) * 8] = a1;
      *(bf16x8*)&Bs[buf][(c1 >> 2) * 40 + (c1 & 3) * 8] = b1;
      if (k0 + 32 < K) {
        a0 = *(const bf16x8*)&A [(size_t)(m0 + (c0 >> 2)) * K + k0 + 32 + (c0 & 3) * 8];
        b0 = *(const bf16x8*)&Bt[(size_t)(n0 + (c0 >> 2)) * K + k0 + 32 + (c0 & 3) * 8];
        a1 = *(const bf16x8*)&A [(size_t)(m0 + (c1 >> 2)) * K + k0 + 32 + (c1 & 3) * 8];
        b1 = *(const bf16x8*)&Bt[(size_t)(n0 + (c1 >> 2)) * K + k0 + 32 + (c1 & 3) * 8];
      }
      __syncthreads();
#pragma unroll
      for (int mt = 0; mt < 2; ++mt) {
        bf16x16 a = ld_frag_a(&As[buf][(wm * 32 + mt * 16 + lo) * 40], hi);
#pragma unroll
        for (int nt = 0; nt < 4; ++nt) {
          bf16x16 b = ld_frag_b(&Bs[buf][(wn * 64 + nt * 16 + lo) * 40], hi);
          acc[mt][nt] = wmma_bf16(a, b, acc[mt][nt]);
        }
      }
      buf ^= 1;
    }
  }
#endif

  // epilogue: C layout -> row = r + 8*hi, col = lane&15
#pragma unroll
  for (int mt = 0; mt < 2; ++mt) {
    int rb = m0 + wm * 32 + mt * 16 + 8 * hi;
#pragma unroll
    for (int nt = 0; nt < 4; ++nt) {
      int col = n0 + wn * 64 + nt * 16 + lo;
      float bv = bias ? bias[col] : 0.f;
#pragma unroll
      for (int r = 0; r < 8; ++r) {
        size_t idx = (size_t)(rb + r) * N + col;
        float v = acc[mt][nt][r] + bv;
        if (residual) v += residual[idx];
        if (act_gelu) v = 0.5f * v * (1.f + erff(v * 0.70710678118654752f));
        if (outf) outf[idx] = v;
        if (outb) outb[idx] = (bf16_t)v;
      }
    }
  }
}

// --- Flash attention: grid (S/64, B*H), block 128 (4 waves, 16 q-rows each) -
__global__ __launch_bounds__(128) void attn_kernel(const bf16_t* __restrict__ Qg,
                                                   const bf16_t* __restrict__ Kg,
                                                   const bf16_t* __restrict__ Vg,
                                                   bf16_t* __restrict__ Og) {
  __shared__ alignas(16) bf16_t Qs[64 * 136];    // [seq][hd]
  __shared__ alignas(16) bf16_t Ks[64 * 136];    // [seq][hd]
  __shared__ alignas(16) bf16_t Vts[128 * 72];   // [hd][seq]  (transposed)
  __shared__ alignas(16) bf16_t Ps[4 * 16 * 72]; // per-wave P tile [16][64]

  int tid = threadIdx.x, lane = tid & 31, wave = tid >> 5;
  int hi = lane >> 4, lo = lane & 15;
  int qt = blockIdx.x, bh = blockIdx.y;
  int b = bh >> 2, h = bh & 3;

  size_t qbase = ((size_t)b * SEQ + qt * 64) * DMODEL + h * HDIM;
  for (int c = tid; c < 1024; c += 128) {           // 64x128 bf16 Q tile
    int r = c >> 4, kc = (c & 15) * 8;
    *(bf16x8*)&Qs[r * 136 + kc] = *(const bf16x8*)&Qg[qbase + (size_t)r * DMODEL + kc];
  }

  float m_r[8], l_r[8];
  f32x8 o_acc[8];
#pragma unroll
  for (int r = 0; r < 8; ++r) { m_r[r] = -INFINITY; l_r[r] = 0.f; }
#pragma unroll
  for (int t = 0; t < 8; ++t)
#pragma unroll
    for (int r = 0; r < 8; ++r) o_acc[t][r] = 0.f;

  const float scale = 0.08838834764831845f;  // 1/sqrt(128)

  for (int j = 0; j < SEQ; j += 64) {
    __syncthreads();
    size_t kb = ((size_t)b * SEQ + j) * DMODEL + h * HDIM;
    for (int c = tid; c < 1024; c += 128) {
      int r = c >> 4, kc = (c & 15) * 8;
      *(bf16x8*)&Ks[r * 136 + kc] = *(const bf16x8*)&Kg[kb + (size_t)r * DMODEL + kc];
      bf16x8 vv = *(const bf16x8*)&Vg[kb + (size_t)r * DMODEL + kc];
#pragma unroll
      for (int e = 0; e < 8; ++e) Vts[(kc + e) * 72 + r] = vv[e];
      if (j + 64 < SEQ) {  // speculative prefetch of next K/V tile
        __builtin_prefetch(&Kg[kb + (size_t)(r + 64) * DMODEL + kc], 0, 1);
        __builtin_prefetch(&Vg[kb + (size_t)(r + 64) * DMODEL + kc], 0, 1);
      }
    }
    __syncthreads();

    // ---- S = Q K^T : 4 n-tiles x 4 k-chunks = 16 wmma --------------------
    f32x8 sc[4];
#pragma unroll
    for (int nt = 0; nt < 4; ++nt)
#pragma unroll
      for (int r = 0; r < 8; ++r) sc[nt][r] = 0.f;
#pragma unroll
    for (int kk = 0; kk < 4; ++kk) {
      bf16x16 a = ld_frag_a(&Qs[(wave * 16 + lo) * 136 + kk * 32], hi);
#pragma unroll
      for (int nt = 0; nt < 4; ++nt) {
        bf16x16 bfr = ld_frag_b(&Ks[(nt * 16 + lo) * 136 + kk * 32], hi);
        sc[nt] = wmma_bf16(a, bfr, sc[nt]);
      }
    }
#pragma unroll
    for (int nt = 0; nt < 4; ++nt)
#pragma unroll
      for (int r = 0; r < 8; ++r) sc[nt][r] *= scale;

    // ---- online softmax (rows live across 16 lanes, same vgpr index) -----
    float alpha[8];
#pragma unroll
    for (int r = 0; r < 8; ++r) {
      float v = fmaxf(fmaxf(sc[0][r], sc[1][r]), fmaxf(sc[2][r], sc[3][r]));
#pragma unroll
      for (int off = 1; off < 16; off <<= 1) v = fmaxf(v, __shfl_xor(v, off, 16));
      float mn = fmaxf(m_r[r], v);
      float al = __expf(m_r[r] - mn);
      float rs = 0.f;
#pragma unroll
      for (int nt = 0; nt < 4; ++nt) {
        float p = __expf(sc[nt][r] - mn);
        sc[nt][r] = p;
        rs += p;
      }
#pragma unroll
      for (int off = 1; off < 16; off <<= 1) rs += __shfl_xor(rs, off, 16);
      l_r[r] = l_r[r] * al + rs;
      m_r[r] = mn;
      alpha[r] = al;
    }
#pragma unroll
    for (int t = 0; t < 8; ++t)
#pragma unroll
      for (int r = 0; r < 8; ++r) o_acc[t][r] *= alpha[r];

    // ---- P (C-layout f32) -> LDS bf16 in A-layout-consumable form --------
    bf16_t* pw = &Ps[wave * 16 * 72];
#pragma unroll
    for (int nt = 0; nt < 4; ++nt)
#pragma unroll
      for (int r = 0; r < 8; ++r)
        pw[(r + 8 * hi) * 72 + nt * 16 + lo] = (bf16_t)sc[nt][r];

    // ---- O += P V : 8 hd-tiles x 2 k-chunks = 16 wmma --------------------
#pragma unroll
    for (int kk = 0; kk < 2; ++kk) {
      bf16x16 a = ld_frag_a(&pw[lo * 72 + kk * 32], hi);
#pragma unroll
      for (int t = 0; t < 8; ++t) {
        bf16x16 bfr = ld_frag_b(&Vts[(t * 16 + lo) * 72 + kk * 32], hi);
        o_acc[t] = wmma_bf16(a, bfr, o_acc[t]);
      }
    }
  }

  // ---- normalize + write ctx as [token, h*128 + d] bf16 ------------------
  size_t ob = ((size_t)b * SEQ + qt * 64 + wave * 16) * DMODEL + h * HDIM;
#pragma unroll
  for (int t = 0; t < 8; ++t)
#pragma unroll
    for (int r = 0; r < 8; ++r) {
      float val = o_acc[t][r] / l_r[r];
      Og[ob + (size_t)(r + 8 * hi) * DMODEL + t * 16 + lo] = (bf16_t)val;
    }
}

// ---------------------------------------------------------------------------
extern "C" void kernel_launch(void* const* d_in, const int* in_sizes, int n_in,
                              void* d_out, int out_size, void* d_ws, size_t ws_size,
                              hipStream_t stream) {
  (void)in_sizes; (void)n_in; (void)out_size; (void)ws_size;
  const float* x1    = (const float*)d_in[0];
  const float* x2    = (const float*)d_in[1];
  const float* ln1_g = (const float*)d_in[2];
  const float* ln1_b = (const float*)d_in[3];
  const float* ln2_g = (const float*)d_in[4];
  const float* ln2_b = (const float*)d_in[5];
  const float* lnf_g = (const float*)d_in[6];
  const float* lnf_b = (const float*)d_in[7];

  char* ws = (char*)d_ws;
  size_t off = 0;
  auto carve = [&](size_t bytes) {
    void* p = ws + off;
    off += (bytes + 255) & ~(size_t)255;
    return p;
  };
  const size_t TD = (size_t)NTOK * DMODEL;
  bf16_t* Wt10  = (bf16_t*)carve(10 * (size_t)DMODEL * DMODEL * 2);
  bf16_t* W1t   = (bf16_t*)carve((size_t)DMODEL * MLPD * 2);
  bf16_t* W2t   = (bf16_t*)carve((size_t)MLPD * DMODEL * 2);
  float*  x1n_f = (float*)carve(TD * 4);
  float*  x2n_f = (float*)carve(TD * 4);
  bf16_t* x1n_b = (bf16_t*)carve(TD * 2);
  bf16_t* x2n_b = (bf16_t*)carve(TD * 2);
  bf16_t* qb    = (bf16_t*)carve(TD * 2);
  bf16_t* kb    = (bf16_t*)carve(TD * 2);
  bf16_t* vb    = (bf16_t*)carve(TD * 2);
  bf16_t* ctxb  = (bf16_t*)carve(TD * 2);
  bf16_t* src1b = (bf16_t*)carve(TD * 2);
  bf16_t* src2b = (bf16_t*)carve(TD * 2);
  float*  xf    = (float*)carve(TD * 4);
  bf16_t* hnb   = (bf16_t*)carve(TD * 2);
  bf16_t* h1b   = (bf16_t*)carve((size_t)NTOK * MLPD * 2);

  const int widx[10] = {8, 10, 12, 14, 16, 18, 20, 22, 24, 26};
  for (int i = 0; i < 10; ++i)
    wcvt_kernel<<<(DMODEL * DMODEL + 255) / 256, 256, 0, stream>>>(
        (const float*)d_in[widx[i]], Wt10 + (size_t)i * DMODEL * DMODEL, DMODEL, DMODEL);
  wcvt_kernel<<<(DMODEL * MLPD + 255) / 256, 256, 0, stream>>>(
      (const float*)d_in[28], W1t, DMODEL, MLPD);
  wcvt_kernel<<<(MLPD * DMODEL + 255) / 256, 256, 0, stream>>>(
      (const float*)d_in[30], W2t, MLPD, DMODEL);
  bf16_t* Wq1t  = Wt10 + 0 * (size_t)DMODEL * DMODEL;
  bf16_t* Wk1t  = Wt10 + 1 * (size_t)DMODEL * DMODEL;
  bf16_t* Wv1t  = Wt10 + 2 * (size_t)DMODEL * DMODEL;
  bf16_t* Wq2t  = Wt10 + 3 * (size_t)DMODEL * DMODEL;
  bf16_t* Wk2t  = Wt10 + 4 * (size_t)DMODEL * DMODEL;
  bf16_t* Wv2t  = Wt10 + 5 * (size_t)DMODEL * DMODEL;
  bf16_t* Wq12t = Wt10 + 6 * (size_t)DMODEL * DMODEL;
  bf16_t* Wk12t = Wt10 + 7 * (size_t)DMODEL * DMODEL;
  bf16_t* Wv12t = Wt10 + 8 * (size_t)DMODEL * DMODEL;
  bf16_t* Wot   = Wt10 + 9 * (size_t)DMODEL * DMODEL;

  ln_kernel<<<NTOK, 256, 0, stream>>>(x1, ln1_g, ln1_b, x1n_f, x1n_b);
  ln_kernel<<<NTOK, 256, 0, stream>>>(x2, ln2_g, ln2_b, x2n_f, x2n_b);

  dim3 gD(NTOK / 128, DMODEL / 128);      // (128, 4)
  dim3 gM1(NTOK / 128, MLPD / 128);       // (128, 8)
  dim3 gAtt(SEQ / 64, BB * NHEAD);        // (16, 64)

  // stream 1 self-attention
  gemm_kernel<<<gD, 256, 0, stream>>>(x1n_b, Wq1t, (const float*)d_in[9],  nullptr, nullptr, qb, NTOK, DMODEL, DMODEL, 0);
  gemm_kernel<<<gD, 256, 0, stream>>>(x1n_b, Wk1t, (const float*)d_in[11], nullptr, nullptr, kb, NTOK, DMODEL, DMODEL, 0);
  gemm_kernel<<<gD, 256, 0, stream>>>(x1n_b, Wv1t, (const float*)d_in[13], nullptr, nullptr, vb, NTOK, DMODEL, DMODEL, 0);
  attn_kernel<<<gAtt, 128, 0, stream>>>(qb, kb, vb, ctxb);
  gemm_kernel<<<gD, 256, 0, stream>>>(ctxb, Wot, (const float*)d_in[27], x1n_f, nullptr, src1b, NTOK, DMODEL, DMODEL, 0);

  // stream 2 self-attention
  gemm_kernel<<<gD, 256, 0, stream>>>(x2n_b, Wq2t, (const float*)d_in[15], nullptr, nullptr, qb, NTOK, DMODEL, DMODEL, 0);
  gemm_kernel<<<gD, 256, 0, stream>>>(x2n_b, Wk2t, (const float*)d_in[17], nullptr, nullptr, kb, NTOK, DMODEL, DMODEL, 0);
  gemm_kernel<<<gD, 256, 0, stream>>>(x2n_b, Wv2t, (const float*)d_in[19], nullptr, nullptr, vb, NTOK, DMODEL, DMODEL, 0);
  attn_kernel<<<gAtt, 128, 0, stream>>>(qb, kb, vb, ctxb);
  gemm_kernel<<<gD, 256, 0, stream>>>(ctxb, Wot, (const float*)d_in[27], x2n_f, nullptr, src2b, NTOK, DMODEL, DMODEL, 0);

  // cross attention (q from src1, k/v from src2)
  gemm_kernel<<<gD, 256, 0, stream>>>(src1b, Wq12t, (const float*)d_in[21], nullptr, nullptr, qb, NTOK, DMODEL, DMODEL, 0);
  gemm_kernel<<<gD, 256, 0, stream>>>(src2b, Wk12t, (const float*)d_in[23], nullptr, nullptr, kb, NTOK, DMODEL, DMODEL, 0);
  gemm_kernel<<<gD, 256, 0, stream>>>(src2b, Wv12t, (const float*)d_in[25], nullptr, nullptr, vb, NTOK, DMODEL, DMODEL, 0);
  attn_kernel<<<gAtt, 128, 0, stream>>>(qb, kb, vb, ctxb);
  gemm_kernel<<<gD, 256, 0, stream>>>(ctxb, Wot, (const float*)d_in[27], x1, xf, nullptr, NTOK, DMODEL, DMODEL, 0);

  // final LN + MLP
  ln_kernel<<<NTOK, 256, 0, stream>>>(xf, lnf_g, lnf_b, nullptr, hnb);
  gemm_kernel<<<gM1, 256, 0, stream>>>(hnb, W1t, (const float*)d_in[29], nullptr, nullptr, h1b, NTOK, MLPD, DMODEL, 1);
  gemm_kernel<<<gD, 256, 0, stream>>>(h1b, W2t, (const float*)d_in[31], xf, (float*)d_out, nullptr, NTOK, DMODEL, MLPD, 0);
}
